// EncoderLayer_20675972563668
// MI455X (gfx1250) — matrix operand order repaired and checked by reading, not compiled, not run on hardware
//
#include <hip/hip_runtime.h>
#include <hip/hip_bf16.h>

// ---------------------------------------------------------------------------
// Problem constants (match reference)
// ---------------------------------------------------------------------------
constexpr int D_MODEL = 1024;
constexpr int N_HEADS = 16;
constexpr int D_K     = 64;
constexpr int D_FF    = 4096;
constexpr int BATCH   = 2;
constexpr int SEQ     = 2048;
constexpr int MS      = BATCH * SEQ;          // 4096 total rows

// GEMM tiling
constexpr int BM    = 128;   // block tile M
constexpr int BN    = 128;   // block tile N
constexpr int BK    = 64;    // K step staged in LDS (2 WMMA sub-steps)
constexpr int LDPAD = 72;    // padded LDS row length (elements) to break bank conflicts

// ---------------------------------------------------------------------------
// WMMA types / helpers (CDNA5 gfx1250, wave32)
// ---------------------------------------------------------------------------
typedef __attribute__((ext_vector_type(16))) __bf16          v16bf;
typedef __attribute__((ext_vector_type(8)))  float           v8f;
typedef __attribute__((ext_vector_type(8)))  unsigned short  us8;

union FragU {
    v16bf v;
    us8   h[2];
};

__device__ __forceinline__ unsigned short f2bf(float f) {
    union { float f; unsigned int u; } x;
    x.f = f;
    unsigned int u = x.u;
    u += 0x7FFFu + ((u >> 16) & 1u);      // round-to-nearest-even
    return (unsigned short)(u >> 16);
}

#define WMMA_BF16(a, b, c) \
    __builtin_amdgcn_wmma_f32_16x16x32_bf16(false, (a), false, (b), (short)0, (c), false, false)

// A fragment (16x32 bf16): M over lanes, K split 0..7/16..23 (lanes 0-15) vs
// 8..15/24..31 (lanes 16-31).  Source row-major, lda in elements.
__device__ __forceinline__ v16bf load_a_frag(const unsigned short* A, int lda,
                                             int m0, int k0, int lane) {
    int row = m0 + (lane & 15);
    int kb  = k0 + ((lane >> 4) << 3);
    const unsigned short* p = A + (size_t)row * lda + kb;
    FragU f;
    f.h[0] = *(const us8*)(p);
    f.h[1] = *(const us8*)(p + 16);
    return f.v;
}

// B fragment (32x16 bf16): N over lanes; lanes 0-15 hold K=0..7/16..23, lanes 16-31
// K=8..15/24..31.  Source stored "N-major, K contiguous" with leading dim ldb.
__device__ __forceinline__ v16bf load_bt_frag(const unsigned short* Bt, int ldb,
                                              int n0, int k0, int lane) {
    int col = n0 + (lane & 15);
    int ks  = k0 + ((lane >> 4) << 3);
    const unsigned short* p = Bt + (size_t)col * ldb + ks;
    FragU f;
    f.h[0] = *(const us8*)(p);
    f.h[1] = *(const us8*)(p + 16);
    return f.v;
}

// ---------------------------------------------------------------------------
// CDNA5 async global->LDS copy (ASYNCcnt-tracked).  The LDS operand is the
// low 32 bits of the flat shared-space address (= LDS offset per ISA 10.2).
// ---------------------------------------------------------------------------
__device__ __forceinline__ void async_ld_b128(const unsigned short* lds_dst,
                                              const unsigned short* gsrc) {
    unsigned lds_off = (unsigned)(size_t)lds_dst;
    asm volatile("global_load_async_to_lds_b128 %0, %1, off"
                 :: "v"(lds_off), "v"(gsrc)
                 : "memory");
}
__device__ __forceinline__ void wait_async0() {
    asm volatile("s_wait_asynccnt 0x0" ::: "memory");
}

// ---------------------------------------------------------------------------
// Weight prep: fp32 W[K][N] -> bf16 transposed Wt[N][K]
// ---------------------------------------------------------------------------
__global__ __launch_bounds__(256) void wprep_kernel(const float* __restrict__ W,
                                                    unsigned short* __restrict__ Wt,
                                                    int K, int N) {
    size_t idx = (size_t)blockIdx.x * 256 + threadIdx.x;   // over N*K
    int n  = (int)(idx / K);
    int kk = (int)(idx % K);
    Wt[idx] = f2bf(W[(size_t)kk * N + n]);
}

// ---------------------------------------------------------------------------
// LayerNorm: fp32 in -> bf16 out (one 1024-wide row per block)
// ---------------------------------------------------------------------------
__global__ __launch_bounds__(256) void ln_kernel(const float* __restrict__ x,
                                                 const float* __restrict__ g,
                                                 const float* __restrict__ bta,
                                                 unsigned short* __restrict__ out) {
    __shared__ float red[8];
    int row = blockIdx.x;
    int tid = threadIdx.x;
    const float* xr = x + (size_t)row * D_MODEL;

    float v[4];
    float s = 0.0f;
#pragma unroll
    for (int i = 0; i < 4; ++i) { v[i] = xr[tid + i * 256]; s += v[i]; }
#pragma unroll
    for (int off = 1; off < 32; off <<= 1) s += __shfl_xor(s, off, 32);
    if ((tid & 31) == 0) red[tid >> 5] = s;
    __syncthreads();
    s = 0.0f;
#pragma unroll
    for (int i = 0; i < 8; ++i) s += red[i];
    float mu = s * (1.0f / D_MODEL);
    __syncthreads();

    float vs = 0.0f;
#pragma unroll
    for (int i = 0; i < 4; ++i) { float d = v[i] - mu; vs += d * d; }
#pragma unroll
    for (int off = 1; off < 32; off <<= 1) vs += __shfl_xor(vs, off, 32);
    if ((tid & 31) == 0) red[tid >> 5] = vs;
    __syncthreads();
    vs = 0.0f;
#pragma unroll
    for (int i = 0; i < 8; ++i) vs += red[i];
    float rs = rsqrtf(vs * (1.0f / D_MODEL) + 1e-5f);

#pragma unroll
    for (int i = 0; i < 4; ++i) {
        int c = tid + i * 256;
        out[(size_t)row * D_MODEL + c] = f2bf((v[i] - mu) * rs * g[c] + bta[c]);
    }
}

// ---------------------------------------------------------------------------
// Generic bf16 WMMA GEMM: C[M,N] = A[M,K] @ Bt[N,K]^T + bias
//   MODE 0: bf16 out           MODE 1: relu -> bf16 out
//   MODE 2: fp32 out = acc + bias + R   MODE 3: bf16 out, V-transpose layout
//
// Block tile 128x128, K-step 64 double-buffered in LDS via CDNA5 async
// global->LDS copies.  8 waves; each wave computes 32(M) x 64(N) = 2x4 WMMA
// fragments, 8 v_wmma per K=32 sub-step, fragments read from LDS.
// ---------------------------------------------------------------------------
template <int MODE>
__device__ __forceinline__ void store_frag(v8f c, int m0, int n0, int lane,
                                           const float* __restrict__ bias,
                                           const float* __restrict__ R,
                                           float* __restrict__ Cf,
                                           unsigned short* __restrict__ Ch,
                                           int N, int Svt) {
    int n = n0 + (lane & 15);
    float bv = bias[n];
#pragma unroll
    for (int r = 0; r < 8; ++r) {
        int m = m0 + r + ((lane >> 4) << 3);
        float val = c[r] + bv;
        if constexpr (MODE == 0) {
            Ch[(size_t)m * N + n] = f2bf(val);
        } else if constexpr (MODE == 1) {
            Ch[(size_t)m * N + n] = f2bf(fmaxf(val, 0.0f));
        } else if constexpr (MODE == 2) {
            size_t i = (size_t)m * N + n;
            Cf[i] = val + R[i];
        } else {  // MODE 3: vt[(b*D_MODEL + n)*SEQ + s]
            int bb = m / Svt;
            int ss = m - bb * Svt;
            Ch[((size_t)bb * N + n) * (size_t)Svt + ss] = f2bf(val);
        }
    }
}

template <int MODE>
__global__ __launch_bounds__(256) void gemm_bf16_kernel(
    const unsigned short* __restrict__ A, const unsigned short* __restrict__ Bt,
    const float* __restrict__ bias, const float* __restrict__ R,
    float* __restrict__ Cf, unsigned short* __restrict__ Ch,
    int M, int N, int K, int Svt) {
    __shared__ unsigned short sA[2][BM * LDPAD];
    __shared__ unsigned short sB[2][BN * LDPAD];

    int tid  = threadIdx.x;
    int lane = tid & 31;
    int w    = tid >> 5;
    int wm   = w & 3;            // 4 M-strips of 32 rows
    int wn   = w >> 2;           // 2 N-strips of 64 cols
    int mblk0 = blockIdx.y * BM;
    int nblk0 = blockIdx.x * BN;

    // stage A[mblk0..+128)[k0..+64) and Bt[nblk0..+128)[k0..+64) into LDS buf
    auto stage = [&](int buf, int k0) {
#pragma unroll
        for (int i = 0; i < 4; ++i) {
            int idx = tid + i * 256;          // 0..1023
            int row = idx >> 3;               // 0..127
            int c   = (idx & 7) << 3;         // 0,8,...,56 (elements)
            async_ld_b128(&sA[buf][row * LDPAD + c],
                          A + (size_t)(mblk0 + row) * K + k0 + c);
            async_ld_b128(&sB[buf][row * LDPAD + c],
                          Bt + (size_t)(nblk0 + row) * K + k0 + c);
        }
    };

    v8f acc[2][4];
#pragma unroll
    for (int mi = 0; mi < 2; ++mi)
#pragma unroll
        for (int ni = 0; ni < 4; ++ni) acc[mi][ni] = (v8f){};

    stage(0, 0);
    wait_async0();
    __syncthreads();

    int cur = 0;
    for (int k0 = 0; k0 < K; k0 += BK) {
        if (k0 + BK < K) stage(cur ^ 1, k0 + BK);

        const unsigned short* la = &sA[cur][0];
        const unsigned short* lb = &sB[cur][0];
#pragma unroll
        for (int ks = 0; ks < BK; ks += 32) {
            v16bf a0 = load_a_frag(la, LDPAD, wm * 32,      ks, lane);
            v16bf a1 = load_a_frag(la, LDPAD, wm * 32 + 16, ks, lane);
            v16bf b0 = load_bt_frag(lb, LDPAD, wn * 64,      ks, lane);
            v16bf b1 = load_bt_frag(lb, LDPAD, wn * 64 + 16, ks, lane);
            v16bf b2 = load_bt_frag(lb, LDPAD, wn * 64 + 32, ks, lane);
            v16bf b3 = load_bt_frag(lb, LDPAD, wn * 64 + 48, ks, lane);
            acc[0][0] = WMMA_BF16(a0, b0, acc[0][0]);
            acc[0][1] = WMMA_BF16(a0, b1, acc[0][1]);
            acc[0][2] = WMMA_BF16(a0, b2, acc[0][2]);
            acc[0][3] = WMMA_BF16(a0, b3, acc[0][3]);
            acc[1][0] = WMMA_BF16(a1, b0, acc[1][0]);
            acc[1][1] = WMMA_BF16(a1, b1, acc[1][1]);
            acc[1][2] = WMMA_BF16(a1, b2, acc[1][2]);
            acc[1][3] = WMMA_BF16(a1, b3, acc[1][3]);
        }

        wait_async0();     // our prefetch of next buffer has landed
        __syncthreads();   // everyone done reading cur, next buffer visible
        cur ^= 1;
    }

    int mblk = mblk0 + wm * 32;
    int nblk = nblk0 + wn * 64;
#pragma unroll
    for (int mi = 0; mi < 2; ++mi)
#pragma unroll
        for (int ni = 0; ni < 4; ++ni)
            store_frag<MODE>(acc[mi][ni], mblk + mi * 16, nblk + ni * 16, lane,
                             bias, R, Cf, Ch, N, Svt);
}

// ---------------------------------------------------------------------------
// Flash attention: one wave owns 16 q rows of one (b,h).  Online softmax over
// 32-key blocks; scores and P@V both via WMMA; P re-laid out C->A through LDS.
//   q,k: bf16 (B,S,D) layout.  vt: bf16 [(b*D + h*64 + d)][s].  ctx: (B,S,D).
// ---------------------------------------------------------------------------
__global__ __launch_bounds__(256) void flash_kernel(
    const unsigned short* __restrict__ q, const unsigned short* __restrict__ k,
    const unsigned short* __restrict__ vt, const unsigned char* __restrict__ mask,
    unsigned short* __restrict__ ctx) {
    __shared__ unsigned short plds[8][16 * 32];

    int lane = threadIdx.x & 31;
    int w    = threadIdx.x >> 5;
    int h    = blockIdx.y;
    int b    = blockIdx.z;
    int q0   = (blockIdx.x * 8 + w) * 16;

    const unsigned short* qbase = q  + (size_t)b * SEQ * D_MODEL + h * D_K;
    const unsigned short* kbase = k  + (size_t)b * SEQ * D_MODEL + h * D_K;
    const unsigned short* vbase = vt + ((size_t)b * D_MODEL + h * D_K) * SEQ;
    const unsigned char*  mb    = mask + (size_t)b * SEQ;

    // Q fragments held for the whole key loop (Dk=64 -> two K=32 A-frags)
    v16bf qf0 = load_a_frag(qbase, D_MODEL, q0, 0,  lane);
    v16bf qf1 = load_a_frag(qbase, D_MODEL, q0, 32, lane);

    float mrow[8], lrow[8];
    v8f o0 = {}, o1 = {}, o2 = {}, o3 = {};
#pragma unroll
    for (int r = 0; r < 8; ++r) { mrow[r] = -3.0e38f; lrow[r] = 0.0f; }

    unsigned short* pl = &plds[w][0];

    for (int key0 = 0; key0 < SEQ; key0 += 32) {
        // ----- scores S = Q @ K^T (16 x 32), scaled ---------------------
        v8f s0 = {}, s1 = {};
        v16bf kb00 = load_bt_frag(kbase, D_MODEL, key0,      0,  lane);
        v16bf kb01 = load_bt_frag(kbase, D_MODEL, key0,      32, lane);
        v16bf kb10 = load_bt_frag(kbase, D_MODEL, key0 + 16, 0,  lane);
        v16bf kb11 = load_bt_frag(kbase, D_MODEL, key0 + 16, 32, lane);
        s0 = WMMA_BF16(qf0, kb00, s0);
        s0 = WMMA_BF16(qf1, kb01, s0);
        s1 = WMMA_BF16(qf0, kb10, s1);
        s1 = WMMA_BF16(qf1, kb11, s1);

        bool msk0 = mb[key0 + (lane & 15)] != 0;
        bool msk1 = mb[key0 + 16 + (lane & 15)] != 0;

        // ----- online softmax + stage P (bf16) into LDS in (row,col) ----
#pragma unroll
        for (int r = 0; r < 8; ++r) {
            float sv0 = msk0 ? -3.0e38f : s0[r] * 0.125f;   // 1/sqrt(64)
            float sv1 = msk1 ? -3.0e38f : s1[r] * 0.125f;
            float mx  = fmaxf(sv0, sv1);
#pragma unroll
            for (int off = 1; off < 16; off <<= 1) mx = fmaxf(mx, __shfl_xor(mx, off, 32));
            float mnew  = fmaxf(mrow[r], mx);
            float scale = __expf(mrow[r] - mnew);
            float p0 = msk0 ? 0.0f : __expf(sv0 - mnew);
            float p1 = msk1 ? 0.0f : __expf(sv1 - mnew);
            float ps = p0 + p1;
#pragma unroll
            for (int off = 1; off < 16; off <<= 1) ps += __shfl_xor(ps, off, 32);
            lrow[r] = lrow[r] * scale + ps;
            mrow[r] = mnew;
            o0[r] *= scale; o1[r] *= scale; o2[r] *= scale; o3[r] *= scale;

            int row = r + ((lane >> 4) << 3);
            pl[row * 32 +       (lane & 15)] = f2bf(p0);
            pl[row * 32 + 16 +  (lane & 15)] = f2bf(p1);
        }

        // ----- ctx += P (16x32) @ V (32x64) -----------------------------
        v16bf pf  = load_a_frag(pl, 32, 0, 0, lane);   // same-wave DS in-order
        v16bf vf0 = load_bt_frag(vbase, SEQ, 0,  key0, lane);
        v16bf vf1 = load_bt_frag(vbase, SEQ, 16, key0, lane);
        v16bf vf2 = load_bt_frag(vbase, SEQ, 32, key0, lane);
        v16bf vf3 = load_bt_frag(vbase, SEQ, 48, key0, lane);
        o0 = WMMA_BF16(pf, vf0, o0);
        o1 = WMMA_BF16(pf, vf1, o1);
        o2 = WMMA_BF16(pf, vf2, o2);
        o3 = WMMA_BF16(pf, vf3, o3);
    }

    // ----- normalize and store ctx (bf16, (B,S,D) layout) ----------------
#pragma unroll
    for (int r = 0; r < 8; ++r) {
        float inv = 1.0f / lrow[r];
        int m = q0 + r + ((lane >> 4) << 3);
        size_t off = ((size_t)b * SEQ + m) * D_MODEL + h * D_K + (lane & 15);
        ctx[off + 0]  = f2bf(o0[r] * inv);
        ctx[off + 16] = f2bf(o1[r] * inv);
        ctx[off + 32] = f2bf(o2[r] * inv);
        ctx[off + 48] = f2bf(o3[r] * inv);
    }
}

// ---------------------------------------------------------------------------
// Host-side orchestration
// ---------------------------------------------------------------------------
extern "C" void kernel_launch(void* const* d_in, const int* in_sizes, int n_in,
                              void* d_out, int out_size, void* d_ws, size_t ws_size,
                              hipStream_t stream) {
    (void)in_sizes; (void)n_in; (void)out_size; (void)ws_size;

    const float* src  = (const float*)d_in[0];
    const unsigned char* mask = (const unsigned char*)d_in[1];   // jnp.bool_ = 1 byte
    const float* Wq = (const float*)d_in[2];  const float* bq = (const float*)d_in[3];
    const float* Wk = (const float*)d_in[4];  const float* bk = (const float*)d_in[5];
    const float* Wv = (const float*)d_in[6];  const float* bv = (const float*)d_in[7];
    const float* Wo = (const float*)d_in[8];  const float* bo = (const float*)d_in[9];
    const float* ln1g = (const float*)d_in[10]; const float* ln1b = (const float*)d_in[11];
    const float* ln2g = (const float*)d_in[12]; const float* ln2b = (const float*)d_in[13];
    const float* W1 = (const float*)d_in[14]; const float* b1 = (const float*)d_in[15];
    const float* W2 = (const float*)d_in[16]; const float* b2 = (const float*)d_in[17];

    // ---- workspace layout ------------------------------------------------
    char* base = (char*)d_ws;
    size_t off = 0;
    auto alloc = [&](size_t bytes) -> char* {
        char* p = base + off;
        off += (bytes + 255) & ~(size_t)255;
        return p;
    };
    unsigned short* xb   = (unsigned short*)alloc((size_t)MS * D_MODEL * 2);
    unsigned short* qb   = (unsigned short*)alloc((size_t)MS * D_MODEL * 2);
    unsigned short* kb   = (unsigned short*)alloc((size_t)MS * D_MODEL * 2);
    unsigned short* vt   = (unsigned short*)alloc((size_t)MS * D_MODEL * 2);
    unsigned short* ctx  = (unsigned short*)alloc((size_t)MS * D_MODEL * 2);
    unsigned short* yb   = (unsigned short*)alloc((size_t)MS * D_MODEL * 2);
    float*          src2 = (float*)         alloc((size_t)MS * D_MODEL * 4);
    unsigned short* h1   = (unsigned short*)alloc((size_t)MS * D_FF * 2);
    unsigned short* Wqt  = (unsigned short*)alloc((size_t)D_MODEL * D_MODEL * 2);
    unsigned short* Wkt  = (unsigned short*)alloc((size_t)D_MODEL * D_MODEL * 2);
    unsigned short* Wvt  = (unsigned short*)alloc((size_t)D_MODEL * D_MODEL * 2);
    unsigned short* Wot  = (unsigned short*)alloc((size_t)D_MODEL * D_MODEL * 2);
    unsigned short* W1t  = (unsigned short*)alloc((size_t)D_MODEL * D_FF * 2);
    unsigned short* W2t  = (unsigned short*)alloc((size_t)D_FF * D_MODEL * 2);

    // ---- 1. weight conversion + transpose (fp32 -> bf16) -----------------
    int gb_dd = (D_MODEL * D_MODEL) / 256;
    int gb_df = (D_MODEL * D_FF)    / 256;
    wprep_kernel<<<gb_dd, 256, 0, stream>>>(Wq, Wqt, D_MODEL, D_MODEL);
    wprep_kernel<<<gb_dd, 256, 0, stream>>>(Wk, Wkt, D_MODEL, D_MODEL);
    wprep_kernel<<<gb_dd, 256, 0, stream>>>(Wv, Wvt, D_MODEL, D_MODEL);
    wprep_kernel<<<gb_dd, 256, 0, stream>>>(Wo, Wot, D_MODEL, D_MODEL);
    wprep_kernel<<<gb_df, 256, 0, stream>>>(W1, W1t, D_MODEL, D_FF);
    wprep_kernel<<<gb_df, 256, 0, stream>>>(W2, W2t, D_FF, D_MODEL);

    // ---- 2. LN1 -> xb (bf16) ---------------------------------------------
    ln_kernel<<<MS, 256, 0, stream>>>(src, ln1g, ln1b, xb);

    // ---- 3. Q,K,V projections (V stored transposed for flash) -----------
    dim3 gqkv(D_MODEL / BN, MS / BM);     // (8, 32)
    gemm_bf16_kernel<0><<<gqkv, 256, 0, stream>>>(xb, Wqt, bq, nullptr, nullptr, qb,
                                                  MS, D_MODEL, D_MODEL, SEQ);
    gemm_bf16_kernel<0><<<gqkv, 256, 0, stream>>>(xb, Wkt, bk, nullptr, nullptr, kb,
                                                  MS, D_MODEL, D_MODEL, SEQ);
    gemm_bf16_kernel<3><<<gqkv, 256, 0, stream>>>(xb, Wvt, bv, nullptr, nullptr, vt,
                                                  MS, D_MODEL, D_MODEL, SEQ);

    // ---- 4. flash attention -> ctx (bf16) --------------------------------
    dim3 gfa(SEQ / (16 * 8), N_HEADS, BATCH);
    flash_kernel<<<gfa, 256, 0, stream>>>(qb, kb, vt, mask, ctx);

    // ---- 5. output projection + residual -> src2 (fp32) -------------------
    gemm_bf16_kernel<2><<<gqkv, 256, 0, stream>>>(ctx, Wot, bo, src, src2, nullptr,
                                                  MS, D_MODEL, D_MODEL, SEQ);

    // ---- 6. LN2 -> yb (bf16) ---------------------------------------------
    ln_kernel<<<MS, 256, 0, stream>>>(src2, ln2g, ln2b, yb);

    // ---- 7. FFN up + relu -> h1 (bf16) ------------------------------------
    dim3 gff1(D_FF / BN, MS / BM);        // (32, 32)
    gemm_bf16_kernel<1><<<gff1, 256, 0, stream>>>(yb, W1t, b1, nullptr, nullptr, h1,
                                                  MS, D_FF, D_MODEL, SEQ);

    // ---- 8. FFN down + residual -> d_out (fp32) ----------------------------
    gemm_bf16_kernel<2><<<gqkv, 256, 0, stream>>>(h1, W2t, b2, src2, (float*)d_out,
                                                  nullptr, MS, D_MODEL, D_FF, SEQ);
}
